// Tokenizer_81793357185107
// MI455X (gfx1250) — matrix operand order repaired and checked by reading, not compiled, not run on hardware
//
#include <hip/hip_runtime.h>
#include <hip/hip_bf16.h>
#include <stdint.h>

// ---------------------------------------------------------------------------
// Problem constants (match setup_inputs): B=16, C=256, H=W=32 -> HW=1024,
// E=512, V=8192, Npix = 16384.
// ---------------------------------------------------------------------------

typedef __attribute__((ext_vector_type(2)))  float  f32x2;
typedef __attribute__((ext_vector_type(8)))  float  f32x8;
typedef __attribute__((ext_vector_type(4)))  int    i32x4;
typedef __attribute__((ext_vector_type(8)))  __bf16 bf16x8;
typedef __attribute__((ext_vector_type(16))) __bf16 bf16x16;

struct BF16Pair { bf16x8 lo, hi; };

__device__ __forceinline__ bf16x16 join16(bf16x8 lo, bf16x8 hi) {
  BF16Pair p{lo, hi};
  return __builtin_bit_cast(bf16x16, p);
}

// Async global->LDS staging (gfx1250 ASYNCcnt path), guarded so absence of
// the builtins degrades to synchronous register staging. The builtin's
// parameters are typed as <4 x i32> pointers in AS1 (global) / AS3 (LDS).
#if defined(__has_builtin)
#if __has_builtin(__builtin_amdgcn_global_load_async_to_lds_b128) && \
    __has_builtin(__builtin_amdgcn_s_wait_asynccnt)
#define USE_ASYNC_LDS 1
#endif
#endif

typedef i32x4 __attribute__((address_space(1))) as1_i32x4;
typedef i32x4 __attribute__((address_space(3))) as3_i32x4;

__device__ __forceinline__ void stage16(const __bf16* __restrict__ g, __bf16* l) {
#ifdef USE_ASYNC_LDS
  __builtin_amdgcn_global_load_async_to_lds_b128(
      (as1_i32x4*)g, (as3_i32x4*)l, 0, 0);
#else
  *(bf16x8*)l = *(const bf16x8*)g;
#endif
}

__device__ __forceinline__ void stage_wait() {
#ifdef USE_ASYNC_LDS
  __builtin_amdgcn_s_wait_asynccnt(0);
#endif
}

// ---------------------------------------------------------------------------
// Kernel A: normalize embedding rows -> bf16 en  [8192 x 512]
// ---------------------------------------------------------------------------
__global__ void norm_embed_kernel(const float* __restrict__ emb,
                                  __bf16* __restrict__ enb) {
  __shared__ float red[256];
  const int row = blockIdx.x;
  const int t   = threadIdx.x;
  const float* er = emb + (size_t)row * 512;
  const float a = er[t];
  const float b = er[t + 256];
  red[t] = a * a + b * b;
  __syncthreads();
  for (int off = 128; off > 0; off >>= 1) {
    if (t < off) red[t] += red[t + off];
    __syncthreads();
  }
  const float rn = 1.0f / (1e-8f + sqrtf(red[0]));
  __bf16* orow = enb + (size_t)row * 512;
  orow[t]       = (__bf16)(a * rn);
  orow[t + 256] = (__bf16)(b * rn);
}

// ---------------------------------------------------------------------------
// Kernel B: pre-quant 1x1 conv as GEMM via V_WMMA_F32_16X16X4_F32.
// One block = 16 pixels x all 512 output channels. 8 waves, 4 N-tiles each.
// Writes exact fp32 z to d_out ([B,E,H,W] layout) and bf16 zn (row-major
// [16384,512]) to workspace for the sims GEMM.
// ---------------------------------------------------------------------------
__global__ void prequant_kernel(const float* __restrict__ x,
                                const float* __restrict__ w_pre,
                                const float* __restrict__ b_pre,
                                float* __restrict__ z_out,
                                __bf16* __restrict__ znb) {
  __shared__ float xs[16][258];   // 16 pixels x 256 channels, padded
  __shared__ float nrm[16];
  __shared__ float rns[16];

  const int tid = threadIdx.x;
  const int p0  = blockIdx.x * 16;     // tile never crosses batch boundary
  const int b   = p0 >> 10;
  const int hw0 = p0 & 1023;

  if (tid < 16) nrm[tid] = 0.0f;

  // Stage x tile: x[b,c,h,w] -> xs[pixel][c]; lanes vary pixel (contiguous hw).
  {
    const int m   = tid & 15;
    const int c16 = tid >> 4;
    for (int cc = 0; cc < 16; ++cc) {
      const int c = c16 * 16 + cc;
      xs[m][c] = x[((size_t)(b * 256 + c)) * 1024 + hw0 + m];
    }
  }
  __syncthreads();

  const int lane    = tid & 31;
  const int wave    = tid >> 5;
  const int n       = lane & 15;
  const int half    = lane >> 4;
  const int colbase = wave * 64;

  f32x8 acc[4] = {};

#if defined(__has_builtin) && __has_builtin(__builtin_amdgcn_wmma_f32_16x16x4_f32)
  for (int ks = 0; ks < 64; ++ks) {
    const int k0 = ks * 4;
    // A 16x4 f32: lane l -> M = l&15, K = (l>>4)*2 + {0,1}
    f32x2 a;
    a.x = xs[n][k0 + half * 2];
    a.y = xs[n][k0 + half * 2 + 1];
#pragma unroll
    for (int j = 0; j < 4; ++j) {
      const int col = colbase + j * 16 + n;
      // B 4x16 f32: lane l -> N = l&15, K = (l>>4)*2 + {0,1}; B[k][n] = w_pre[n][k]
      const f32x2 bw = *(const f32x2*)(w_pre + (size_t)col * 256 + k0 + half * 2);
      acc[j] = __builtin_amdgcn_wmma_f32_16x16x4_f32(
          false, a, false, bw, (short)0, acc[j], false, false);
    }
  }
#else
  for (int ks = 0; ks < 256; ++ks) {
#pragma unroll
    for (int j = 0; j < 4; ++j) {
      const int col  = colbase + j * 16 + n;
      const float wv = w_pre[(size_t)col * 256 + ks];
#pragma unroll
      for (int r = 0; r < 8; ++r)
        acc[j][r] += xs[half * 8 + r][ks] * wv;
    }
  }
#endif

  // Bias + per-row sum of squares (C layout: lane l vgpr r -> row half*8+r, col lane&15)
  float ss[8] = {0.f, 0.f, 0.f, 0.f, 0.f, 0.f, 0.f, 0.f};
#pragma unroll
  for (int j = 0; j < 4; ++j) {
    const float bias = b_pre[colbase + j * 16 + n];
#pragma unroll
    for (int r = 0; r < 8; ++r) {
      acc[j][r] += bias;
      ss[r] += acc[j][r] * acc[j][r];
    }
  }
#pragma unroll
  for (int r = 0; r < 8; ++r) {
    float v = ss[r];
    v += __shfl_xor(v, 1, 32);
    v += __shfl_xor(v, 2, 32);
    v += __shfl_xor(v, 4, 32);
    v += __shfl_xor(v, 8, 32);
    if (n == 0) atomicAdd(&nrm[half * 8 + r], v);   // ds_add_f32
  }
  __syncthreads();
  if (tid < 16) rns[tid] = 1.0f / (1e-8f + sqrtf(nrm[tid]));
  __syncthreads();

#pragma unroll
  for (int j = 0; j < 4; ++j) {
    const int e = colbase + j * 16 + n;
#pragma unroll
    for (int r = 0; r < 8; ++r) {
      const int m    = half * 8 + r;
      const float zv = acc[j][r];
      z_out[((size_t)(b * 512 + e)) * 1024 + hw0 + m] = zv;          // z (exact fp32)
      znb[((size_t)(p0 + m)) * 512 + e] = (__bf16)(zv * rns[m]);     // zn (bf16)
    }
  }
}

// ---------------------------------------------------------------------------
// Kernel C: fused sims GEMM (bf16 WMMA) + streaming argmax + gather.
// One block = 64 pixels (4 M-tiles). 8 waves: wave-pair (mt = wave>>1) owns an
// M-tile; the pair splits the 512 N-tiles by parity. en column tiles are
// staged through double-buffered LDS with async global->LDS copies (one
// barrier per iteration): per iteration 32 en rows (2 N-tiles, 32 KB) serve
// 128 WMMAs -> ~64 flop per L2 byte, vs 16 in the unstaged version.
// Argmax carried as packed u64: (orderable(f32) << 32) | (0xFFFFFFFF - col).
// ---------------------------------------------------------------------------
__global__ void vq_kernel(const float* __restrict__ emb,
                          const __bf16* __restrict__ znb,
                          const __bf16* __restrict__ enb,
                          float* __restrict__ zq_out,
                          float* __restrict__ dec_out,
                          float* __restrict__ tok_out) {
  __shared__ __bf16 ens[2][32][520];     // padded: row stride % 64 banks == 4
  __shared__ unsigned long long bestS[64];
  __shared__ int toks[64];

  const int tid    = threadIdx.x;
  const int p0     = blockIdx.x * 64;
  const int b      = p0 >> 10;
  const int hw0    = p0 & 1023;
  const int lane   = tid & 31;
  const int wave   = tid >> 5;
  const int n      = lane & 15;
  const int half   = lane >> 4;
  const int mt     = wave >> 1;          // M-tile 0..3
  const int parity = wave & 1;           // N-tile parity within the pair

  if (tid < 64) bestS[tid] = 0ull;

  // Preload A fragments for this wave's M-tile: bf16 A 16x32 layout ->
  // lane l reads row (l&15), 8 bf16 at k0+(l>>4)*8 and at +16.
  bf16x16 afrag[16];
  {
    const __bf16* zr = znb + (size_t)(p0 + mt * 16 + n) * 512;
#pragma unroll
    for (int ks = 0; ks < 16; ++ks) {
      const int base = ks * 32 + half * 8;
      const bf16x8 lo = *(const bf16x8*)(zr + base);
      const bf16x8 hi = *(const bf16x8*)(zr + base + 16);
      afrag[ks] = join16(lo, hi);
    }
  }

  // Cooperative staging of 32 en rows (iteration it covers rows it*32..+31).
  const int srow = tid >> 3;             // 0..31
  const int sseg = tid & 7;              // 0..7 -> 64 bf16 each
  auto stage_tile = [&](int it, int buf) {
    const __bf16* g = enb + ((size_t)(it * 32 + srow)) * 512 + sseg * 64;
    __bf16* l = &ens[buf][srow][sseg * 64];
#pragma unroll
    for (int q = 0; q < 8; ++q) stage16(g + q * 8, l + q * 8);
  };

  stage_tile(0, 0);                      // prologue

  unsigned long long best[8] = {};
  for (int it = 0; it < 256; ++it) {
    stage_wait();
    __syncthreads();                     // buf (it&1) ready; prev compute done
    if (it + 1 < 256) stage_tile(it + 1, (it + 1) & 1);

    // bf16 B 32x16: lane l reads 8 bf16 of en row (col0+(l&15)) at
    // k0 + (l>>4)*8 and at +16 -- served from LDS, conflict-free (stride 260 dwords).
    const __bf16* brow = &ens[it & 1][parity * 16 + n][half * 8];
    f32x8 c = {};
#pragma unroll
    for (int ks = 0; ks < 16; ++ks) {
      const bf16x8 blo = *(const bf16x8*)(brow + ks * 32);
      const bf16x8 bhi = *(const bf16x8*)(brow + ks * 32 + 16);
      const bf16x16 bfrag = join16(blo, bhi);
      c = __builtin_amdgcn_wmma_f32_16x16x32_bf16(
          false, afrag[ks], false, bfrag, (short)0, c, false, false);
    }

    const int col0 = (it * 2 + parity) * 16;
    const unsigned lowpart = 0xFFFFFFFFu - (unsigned)(col0 + n);
#pragma unroll
    for (int r = 0; r < 8; ++r) {
      const unsigned bits = __float_as_uint(c[r]);
      const unsigned ord  = bits ^ ((unsigned)(((int)bits) >> 31) | 0x80000000u);
      const unsigned long long pk = ((unsigned long long)ord << 32) | lowpart;
      if (pk > best[r]) best[r] = pk;
    }
  }

  // Reduce argmax: 16-lane shfl tree (masks stay within the half-group),
  // then a 2-wave ds_max_u64 combine per row.
#pragma unroll
  for (int r = 0; r < 8; ++r) {
    unsigned long long v = best[r];
    unsigned long long o;
    o = __shfl_xor(v, 1, 32); if (o > v) v = o;
    o = __shfl_xor(v, 2, 32); if (o > v) v = o;
    o = __shfl_xor(v, 4, 32); if (o > v) v = o;
    o = __shfl_xor(v, 8, 32); if (o > v) v = o;
    if (n == 0) atomicMax(&bestS[mt * 16 + half * 8 + r], v);
  }
  __syncthreads();

  if (tid < 64) {
    const unsigned long long pk = bestS[tid];
    const int tok = (int)(0xFFFFFFFFu - (unsigned)(pk & 0xFFFFFFFFull));
    toks[tid] = tok;
    tok_out[p0 + tid] = (float)tok;
  }
  __syncthreads();

  // Gather z_q = embedding[token]; decoder_input forward value == z_q exactly.
  {
    const int m  = tid & 63;
    const int eb = tid >> 6;             // 0..3
    const float* emr = emb + (size_t)toks[m] * 512;
    for (int ee = 0; ee < 128; ++ee) {
      const int e    = eb * 128 + ee;
      const float v  = emr[e];
      const size_t i = ((size_t)(b * 512 + e)) * 1024 + hw0 + m;
      zq_out[i]  = v;
      dec_out[i] = v;
    }
  }
}

// ---------------------------------------------------------------------------
// Host launcher
// ---------------------------------------------------------------------------
extern "C" void kernel_launch(void* const* d_in, const int* in_sizes, int n_in,
                              void* d_out, int out_size, void* d_ws, size_t ws_size,
                              hipStream_t stream) {
  (void)in_sizes; (void)n_in; (void)out_size; (void)ws_size;

  const float* x     = (const float*)d_in[0];   // [16,256,32,32]
  const float* w_pre = (const float*)d_in[1];   // [512,256]
  const float* b_pre = (const float*)d_in[2];   // [512]
  const float* emb   = (const float*)d_in[3];   // [8192,512]

  float* out     = (float*)d_out;
  float* z_out   = out;                          // 16*512*1024 = 8388608
  float* zq_out  = out + (size_t)8388608;
  float* dec_out = out + (size_t)16777216;
  float* tok_out = out + (size_t)25165824;       // 16384 tokens (as float)

  __bf16* znb = (__bf16*)d_ws;                                         // 16 MB
  __bf16* enb = (__bf16*)((char*)d_ws + (size_t)16384 * 512 * 2);      //  8 MB

  norm_embed_kernel<<<dim3(8192), dim3(256), 0, stream>>>(emb, enb);
  prequant_kernel<<<dim3(1024), dim3(256), 0, stream>>>(x, w_pre, b_pre, z_out, znb);
  vq_kernel<<<dim3(256), dim3(256), 0, stream>>>(emb, znb, enb, zq_out, dec_out, tok_out);
}
